// HarmonicSpecDiscriminator_28363964023068
// MI455X (gfx1250) — matrix operand order repaired and checked by reading, not compiled, not run on hardware
//
#include <hip/hip_runtime.h>
#include <hip/hip_bf16.h>
#include <math.h>

typedef __attribute__((ext_vector_type(16))) _Float16 v16h;
typedef __attribute__((ext_vector_type(8)))  float    v8f;

#define LRELU_SLOPE 0.1f
#define EPS_F       1.1920928955078125e-07f
#define TWO_PI      6.283185307179586f

// ---- problem geometry ----
#define BATCH        4
#define T_SAMPLES    96000
#define N_FFT        1024
#define HOP          120
#define WINLEN       600
#define WIN_LP       212          // (1024-600)/2
#define NFRAMES      801
#define NFRAMES_PAD  816          // padded to multiple of 16
#define NBINS        513
#define NBINS_PAD    528          // padded to multiple of 16
#define NKF          7
#define MAXKP        864          // largest padded K among all layers

// SHIFTS[k] = 1000*ln(7/(k+1))
__device__ __constant__ float c_shift[NKF] = {
    1945.910149055313f, 1252.7629684953681f, 847.2978603872037f,
    559.6157879354227f, 336.47223662121295f, 154.15067982725836f, 0.0f
};

// ------------------------------------------------------------------
// Build f16 DFT basis: basC[f][k]=cos(2pi f k/1024), basS likewise sin.
// ------------------------------------------------------------------
__global__ void fill_basis(_Float16* __restrict__ basC, _Float16* __restrict__ basS) {
    int idx = blockIdx.x * blockDim.x + threadIdx.x;
    if (idx >= NBINS_PAD * N_FFT) return;
    int f = idx >> 10, k = idx & 1023;
    int m = (f * k) & 1023;
    float ang = TWO_PI * (float)m * (1.0f / 1024.0f);
    basC[idx] = (_Float16)cosf(ang);
    basS[idx] = (_Float16)sinf(ang);
}

// ------------------------------------------------------------------
// Windowed frame extraction with reflect padding -> f16 frames.
// ------------------------------------------------------------------
__global__ void fill_frames(const float* __restrict__ y, _Float16* __restrict__ frames) {
    int idx = blockIdx.x * blockDim.x + threadIdx.x;
    const int total = BATCH * NFRAMES_PAD * N_FFT;
    if (idx >= total) return;
    int k  = idx & 1023;
    int nt = idx >> 10;
    int t  = nt % NFRAMES_PAD;
    int b  = nt / NFRAMES_PAD;
    float val = 0.0f;
    if (t < NFRAMES) {
        int kn = k - WIN_LP;
        if (kn >= 0 && kn < WINLEN) {
            float w = 0.5f * (1.0f - cosf(TWO_PI * (float)kn * (1.0f / (float)WINLEN)));
            int pos = t * HOP + k - (N_FFT / 2);
            if (pos < 0) pos = -pos;                       // reflect
            if (pos >= T_SAMPLES) pos = 2 * T_SAMPLES - 2 - pos;
            val = y[(size_t)b * T_SAMPLES + pos] * w;
        }
    }
    frames[idx] = (_Float16)val;
}

// ------------------------------------------------------------------
// STFT magnitude via WMMA GEMM: M=16 freq bins, N=16 frames, K=1024.
// ------------------------------------------------------------------
__global__ void stft_wmma(const _Float16* __restrict__ frames,
                          const _Float16* __restrict__ basC,
                          const _Float16* __restrict__ basS,
                          float* __restrict__ mag) {
    const int lane = threadIdx.x & 31;
    const int tb = blockIdx.x * 16;      // frame tile base
    const int fb = blockIdx.y * 16;      // freq tile base
    const int b  = blockIdx.z;

    v8f accR = {}, accI = {};
    const int f   = fb + (lane & 15);                 // A-row this lane owns
    const int t   = tb + (lane & 15);                 // B-col this lane owns
    const int klo = (lane < 16) ? 0 : 8;              // table-order k groups, unused (ISA 7.12.2)
    const int khi = (lane < 16) ? 16 : 24;
    const int kb  = (lane < 16) ? 0 : 16;             // B k group
    const _Float16* frm  = frames + ((size_t)b * NFRAMES_PAD + t) * N_FFT;
    const _Float16* rowC = basC + (size_t)f * N_FFT;
    const _Float16* rowS = basS + (size_t)f * N_FFT;

    for (int kc = 0; kc < N_FFT / 32; ++kc) {
        v16h ac, as, bf;
#pragma unroll
        for (int i = 0; i < 16; ++i) {
            int k0 = kc * 32 + kb + i;
            ac[i] = rowC[k0];
            as[i] = rowS[k0];
        }
#pragma unroll
        for (int i = 0; i < 16; ++i)
            bf[i] = frm[kc * 32 + kb + i];

        accR = __builtin_amdgcn_wmma_f32_16x16x32_f16(false, ac, false, bf,
                                                      (short)0, accR, false, false);
        accI = __builtin_amdgcn_wmma_f32_16x16x32_f16(false, as, false, bf,
                                                      (short)0, accI, false, false);
    }

#pragma unroll
    for (int r = 0; r < 8; ++r) {
        int fr = fb + ((lane < 16) ? r : (r + 8));
        int tt = tb + (lane & 15);
        if (fr < NBINS && tt < NFRAMES) {
            float re = accR[r], im = accI[r];
            mag[((size_t)b * NBINS + fr) * NFRAMES + tt] =
                sqrtf(fmaxf(re * re + im * im, EPS_F));
        }
    }
}

// ------------------------------------------------------------------
// Harmonic lowering: fractional frequency shifts of mag -> (B,7,513,801)
// ------------------------------------------------------------------
__global__ void fill_lowered(const float* __restrict__ mag, float* __restrict__ low) {
    const long long total = (long long)BATCH * NKF * NBINS * NFRAMES;
    long long idx = (long long)blockIdx.x * blockDim.x + threadIdx.x;
    if (idx >= total) return;
    int t  = (int)(idx % NFRAMES);
    int f  = (int)((idx / NFRAMES) % NBINS);
    int fk = (int)((idx / ((long long)NFRAMES * NBINS)) % NKF);
    int b  = (int)(idx / ((long long)NFRAMES * NBINS * NKF));
    float s = c_shift[fk];
    int   s0 = (int)floorf(s);
    float fr = s - (float)s0;
    int r0 = f - s0, r1 = r0 - 1;
    float v = 0.0f;
    if (r0 >= 0 && r0 < NBINS) v += (1.0f - fr) * mag[((size_t)b * NBINS + r0) * NFRAMES + t];
    if (fr > 0.0f && r1 >= 0 && r1 < NBINS) v += fr * mag[((size_t)b * NBINS + r1) * NFRAMES + t];
    low[idx] = v;
}

// ------------------------------------------------------------------
// Weight norm -> f16 weights, A-friendly layout wlay[co][Kpad],
// k = ((kh*KW+kw)*Cin + cin).  co in [Cout,32) zero-filled.
// ------------------------------------------------------------------
__global__ void weight_norm(const float* __restrict__ v, const float* __restrict__ g,
                            _Float16* __restrict__ wlay,
                            int Cout, int Cin, int KH, int KW, int Kpad) {
    __shared__ float red[256];
    __shared__ float sc_s;
    const int co  = blockIdx.x;
    const int tid = threadIdx.x;
    const int Ktot = Cin * KH * KW;
    float s = 0.0f;
    if (co < Cout) {
        const float* vc = v + (size_t)co * Ktot;   // OIHW contiguous per out-channel
        for (int i = tid; i < Ktot; i += 256) { float x = vc[i]; s += x * x; }
    }
    red[tid] = s;
    __syncthreads();
    for (int st = 128; st > 0; st >>= 1) {
        if (tid < st) red[tid] += red[tid + st];
        __syncthreads();
    }
    if (tid == 0) sc_s = (co < Cout) ? (g[co] / sqrtf(red[0])) : 0.0f;
    __syncthreads();
    const float sc = sc_s;
    for (int k = tid; k < Kpad; k += 256) {
        float val = 0.0f;
        if (co < Cout && k < Ktot) {
            int cin = k % Cin;
            int rest = k / Cin;
            int kw = rest % KW, kh = rest / KW;
            val = sc * v[(((size_t)co * Cin + cin) * KH + kh) * KW + kw];
        }
        wlay[(size_t)co * Kpad + k] = (_Float16)val;
    }
}

// ------------------------------------------------------------------
// Async gather of one 32k x 32n B chunk into LDS (f32), exactly 8
// global_load_async_to_lds_b32 per lane (OOB lanes target a trash slot
// so ASYNCcnt advances deterministically; their real slot is zeroed
// with a plain LDS store on a disjoint address).
// ------------------------------------------------------------------
__device__ __forceinline__ void stage_chunk_async(
    const float* __restrict__ in, const int* sBase, const short* sKw,
    float (*sBuf)[36], float* sTrash,
    int kc, int wbase, int strideW, int Win, int n, int kgrp) {
#pragma unroll
    for (int i = 0; i < 8; ++i) {
        int kglob = kc * 32 + kgrp + i;
        int base  = sBase[kglob];
        int w_in  = wbase + n * strideW + (int)sKw[kglob];
        bool valid = (base >= 0) && (w_in >= 0) && (w_in < Win);
        const float* gp = valid ? (in + (size_t)base + w_in) : in;
        float*       lp = valid ? &sBuf[n][kgrp + i]         : &sTrash[n];
        unsigned loff = (unsigned)(size_t)(void*)lp;   // low 32 bits = LDS offset
        asm volatile("global_load_async_to_lds_b32 %0, %1, off"
                     :: "v"(loff), "v"((unsigned long long)(size_t)gp)
                     : "memory");
        if (!valid) sBuf[n][kgrp + i] = 0.0f;
    }
}

// ------------------------------------------------------------------
// Generic conv2d (strideH==1) as implicit GEMM on WMMA.
// 128 threads = 4 waves; tile = 32 co x 32 out-cols at one (b,h).
// B patches gathered straight to LDS with async loads, double-buffered
// against the WMMA; A from f16 weights in [co][K] layout (b128 loads).
// ------------------------------------------------------------------
__global__ void conv2d_wmma(const float* __restrict__ in,
                            const _Float16* __restrict__ wlay,
                            const float* __restrict__ bias,
                            float* __restrict__ out, float* __restrict__ out2,
                            int Cin, int Hin, int Win,
                            int Cout, int Hout, int Wout,
                            int KH, int KW, int strideW, int padH, int padW,
                            int Ktot, int Kpad, int do_lrelu) {
    __shared__ __align__(16) float sBf[2][32][36];  // [buf][n][k], 16B-aligned rows
    __shared__ int   sBase[MAXKP];                  // per-k input row base (or -1)
    __shared__ short sKw[MAXKP];                    // per-k kernel-column offset
    __shared__ float sTrash[32];                    // async sink for OOB lanes

    const int b   = blockIdx.z;
    const int h   = blockIdx.y;
    const int w0  = blockIdx.x * 32;
    const int tid  = threadIdx.x;
    const int lane = tid & 31;
    const int wv   = tid >> 5;
    const int mi   = wv >> 1;           // which 16-row co block
    const int ni   = wv & 1;            // which 16-col block

    // ---- precompute per-k gather bases (hoists div/mod out of hot loop) ----
    for (int k = tid; k < Kpad; k += 128) {
        int base = -1; short skw = 0;
        if (k < Ktot) {
            int cin = k % Cin;
            int rest = k / Cin;
            int kw = rest % KW, kh = rest / KW;
            int h_in = h - padH + kh;               // strideH == 1 everywhere here
            if (h_in >= 0 && h_in < Hin)
                base = ((b * Cin + cin) * Hin + h_in) * Win;
            skw = (short)kw;
        }
        sBase[k] = base;
        sKw[k]   = skw;
    }
    __syncthreads();

    v8f acc = {};
    const int nk    = Kpad >> 5;
    const int n_st  = tid & 31;          // column this thread stages
    const int kgrp  = (tid >> 5) * 8;    // k sub-group this thread stages
    const int wbase = w0 * strideW - padW;

    const int co_a = mi * 16 + (lane & 15);
    const int klo  = (lane < 16) ? 0 : 8;
    const int khi  = (lane < 16) ? 16 : 24;
    const int kb   = (lane < 16) ? 0 : 16;
    const int ncol = ni * 16 + (lane & 15);
    const _Float16* arow = wlay + (size_t)co_a * Kpad;

    stage_chunk_async(in, sBase, sKw, sBf[0], sTrash, 0, wbase, strideW, Win, n_st, kgrp);

    for (int kc = 0; kc < nk; ++kc) {
        if (kc + 1 < nk) {
            __builtin_prefetch(arow + (kc + 1) * 32, 0, 1);   // next A chunk
            stage_chunk_async(in, sBase, sKw, sBf[(kc + 1) & 1], sTrash,
                              kc + 1, wbase, strideW, Win, n_st, kgrp);
            // 8 newer async loads in flight; drain this chunk's 8 (in-order).
            asm volatile("s_wait_asynccnt 0x8" ::: "memory");
        } else {
            asm volatile("s_wait_asynccnt 0x0" ::: "memory");
        }
        __syncthreads();

        // ---- A fragment: one contiguous 32B run per lane ----
        v16h afrag, bfrag;
        const _Float16* ap = arow + kc * 32;
#pragma unroll
        for (int i = 0; i < 16; ++i)
            afrag[i] = ap[kb + i];
        // ---- B fragment: 64B contiguous LDS run per lane, cvt f32->f16 ----
        const float* rp = &sBf[kc & 1][ncol][kb];
#pragma unroll
        for (int i = 0; i < 16; ++i)
            bfrag[i] = (_Float16)rp[i];

        acc = __builtin_amdgcn_wmma_f32_16x16x32_f16(false, afrag, false, bfrag,
                                                     (short)0, acc, false, false);
        __syncthreads();   // protect buf (kc&1) before it is refilled
    }

    // ---- epilogue: bias + leaky relu, masked store ----
    const int w = w0 + ncol;
#pragma unroll
    for (int r = 0; r < 8; ++r) {
        int m  = (lane < 16) ? r : (r + 8);
        int co = mi * 16 + m;
        if (co < Cout && w < Wout) {
            float o = acc[r] + bias[co];
            if (do_lrelu) o = (o > 0.0f) ? o : LRELU_SLOPE * o;
            size_t oi = ((size_t)(b * Cout + co) * Hout + h) * Wout + w;
            out[oi] = o;
            if (out2) out2[oi] = o;     // final layer mirrors into `flat`
        }
    }
}

// ------------------------------------------------------------------
extern "C" void kernel_launch(void* const* d_in, const int* in_sizes, int n_in,
                              void* d_out, int out_size, void* d_ws, size_t ws_size,
                              hipStream_t stream) {
    (void)in_sizes; (void)n_in; (void)out_size; (void)ws_size;
    const float* y  = (const float*)d_in[0];
    const float* vw[6] = { (const float*)d_in[1], (const float*)d_in[4],
                           (const float*)d_in[7], (const float*)d_in[10],
                           (const float*)d_in[13], (const float*)d_in[16] };
    const float* gw[6] = { (const float*)d_in[2], (const float*)d_in[5],
                           (const float*)d_in[8], (const float*)d_in[11],
                           (const float*)d_in[14], (const float*)d_in[17] };
    const float* bw[6] = { (const float*)d_in[3], (const float*)d_in[6],
                           (const float*)d_in[9], (const float*)d_in[12],
                           (const float*)d_in[15], (const float*)d_in[18] };

    // ---- workspace carve-out ----
    char* wsb = (char*)d_ws;
    size_t off = 0;
    auto carve = [&](size_t bytes) -> void* {
        void* p = wsb + off;
        off = (off + bytes + 255) & ~(size_t)255;
        return p;
    };
    _Float16* basC   = (_Float16*)carve((size_t)NBINS_PAD * N_FFT * 2);
    _Float16* basS   = (_Float16*)carve((size_t)NBINS_PAD * N_FFT * 2);
    _Float16* frames = (_Float16*)carve((size_t)BATCH * NFRAMES_PAD * N_FFT * 2);
    float*    mag    = (float*)carve((size_t)BATCH * NBINS * NFRAMES * 4);
    float*    low    = (float*)carve((size_t)BATCH * NKF * NBINS * NFRAMES * 4);
    _Float16* wlay[6];
    const int Kpad_l[6] = { 64, 864, 864, 864, 288, 288 };
    for (int l = 0; l < 6; ++l) wlay[l] = (_Float16*)carve((size_t)32 * Kpad_l[l] * 2);

    // ---- output carve-out (return order: flat, fmap0..fmap5) ----
    float* o    = (float*)d_out;
    float* flat = o;                                   // 4*1*513*101
    float* f0   = o + 207252;                          // 4*32*513*801
    float* f1   = f0 + (size_t)52596864;               // 4*32*513*401
    float* f2   = f1 + (size_t)26331264;               // 4*32*513*201
    float* f3   = f2 + (size_t)13198464;               // 4*32*513*101
    float* f4   = f3 + (size_t)6632064;                // 4*32*513*101
    float* f5   = f4 + (size_t)6632064;                // 4*1*513*101

    // ---- 1) DFT basis + windowed frames ----
    {
        int n = NBINS_PAD * N_FFT;
        fill_basis<<<(n + 255) / 256, 256, 0, stream>>>(basC, basS);
        n = BATCH * NFRAMES_PAD * N_FFT;
        fill_frames<<<(n + 255) / 256, 256, 0, stream>>>(y, frames);
    }
    // ---- 2) STFT magnitude via WMMA ----
    stft_wmma<<<dim3((NFRAMES + 15) / 16, (NBINS + 15) / 16, BATCH), 32, 0, stream>>>(
        frames, basC, basS, mag);
    // ---- 3) harmonic lowering ----
    {
        long long n = (long long)BATCH * NKF * NBINS * NFRAMES;
        fill_lowered<<<(unsigned)((n + 255) / 256), 256, 0, stream>>>(mag, low);
    }
    // ---- 4) weight norm for all layers ----
    weight_norm<<<32, 256, 0, stream>>>(vw[0], gw[0], wlay[0], 32,  7, 1, 7,  64);
    weight_norm<<<32, 256, 0, stream>>>(vw[1], gw[1], wlay[1], 32, 32, 3, 9, 864);
    weight_norm<<<32, 256, 0, stream>>>(vw[2], gw[2], wlay[2], 32, 32, 3, 9, 864);
    weight_norm<<<32, 256, 0, stream>>>(vw[3], gw[3], wlay[3], 32, 32, 3, 9, 864);
    weight_norm<<<32, 256, 0, stream>>>(vw[4], gw[4], wlay[4], 32, 32, 3, 3, 288);
    weight_norm<<<32, 256, 0, stream>>>(vw[5], gw[5], wlay[5],  1, 32, 3, 3, 288);

    // ---- 5) conv stack ----
    conv2d_wmma<<<dim3(26, 513, BATCH), 128, 0, stream>>>(
        low, wlay[0], bw[0], f0, nullptr,
        7, 513, 801,  32, 513, 801,  1, 7, 1, 0, 3,  49, 64, 1);
    conv2d_wmma<<<dim3(13, 513, BATCH), 128, 0, stream>>>(
        f0, wlay[1], bw[1], f1, nullptr,
        32, 513, 801,  32, 513, 401,  3, 9, 2, 1, 4,  864, 864, 1);
    conv2d_wmma<<<dim3(7, 513, BATCH), 128, 0, stream>>>(
        f1, wlay[2], bw[2], f2, nullptr,
        32, 513, 401,  32, 513, 201,  3, 9, 2, 1, 4,  864, 864, 1);
    conv2d_wmma<<<dim3(4, 513, BATCH), 128, 0, stream>>>(
        f2, wlay[3], bw[3], f3, nullptr,
        32, 513, 201,  32, 513, 101,  3, 9, 2, 1, 4,  864, 864, 1);
    conv2d_wmma<<<dim3(4, 513, BATCH), 128, 0, stream>>>(
        f3, wlay[4], bw[4], f4, nullptr,
        32, 513, 101,  32, 513, 101,  3, 3, 1, 1, 1,  288, 288, 1);
    conv2d_wmma<<<dim3(4, 513, BATCH), 128, 0, stream>>>(
        f4, wlay[5], bw[5], f5, flat,
        32, 513, 101,  1, 513, 101,  3, 3, 1, 1, 1,  288, 288, 0);
}
